// SwitchSAE_4973572129208
// MI455X (gfx1250) — compile-verified, matrix operand order checked by reading, not run on hardware
//
#include <hip/hip_runtime.h>

typedef __attribute__((ext_vector_type(16))) __bf16 v16bf;
typedef __attribute__((ext_vector_type(8)))  float  v8f;

#define NTOK   65536
#define DDIM   256
#define NEXP   8
#define MTILE  128
#define LSTRIDE 264   // ushort elems per row: 16B-aligned (264*2%16==0), bank-spread (132 dw %64)
#define TILES_PER_E (NTOK / MTILE)   // 512

typedef unsigned short u16;
typedef unsigned int   u32;

// ---------- helpers ----------
__device__ __forceinline__ u16 f2bf(float f) {
    u32 u = __builtin_bit_cast(u32, f);
    u32 r = u + 0x7FFFu + ((u >> 16) & 1u);   // round-to-nearest-even
    return (u16)(r >> 16);
}

struct U32x8 { uint4 lo; uint4 hi; };

__device__ __forceinline__ v16bf load_frag(const u16* p) {
    U32x8 s;
    s.lo = *(const uint4*)(p);        // K 0..7  of chunk (8 bf16, 16B)
    s.hi = *(const uint4*)(p + 16);   // K 16..23 of chunk
    return __builtin_bit_cast(v16bf, s);
}

// ---------- ws layout (bytes) ----------
#define WS_ENC   0u                       // u16  [8*256*256]
#define WS_DEC   (1u*1024u*1024u)         // u16  [8*256*256]
#define WS_IDX   (2u*1024u*1024u)         // int  [65536]
#define WS_MAXP  (WS_IDX  + 262144u)      // f32  [65536]
#define WS_SORT  (WS_MAXP + 262144u)      // int  [65536]
#define WS_CNT   (WS_SORT + 262144u)      // int  [8]
#define WS_OFF   (WS_CNT + 32u)           // int  [8]
#define WS_CUR   (WS_CNT + 64u)           // int  [8]
#define WS_PSUM  (WS_CNT + 96u)           // f32  [8]

// ---------- d_out layout (floats) ----------
#define O_REC    0u                       // [65536*256]
#define O_LAT    16777216u                // [65536*256]
#define O_WA     33554432u                // [8*256]
#define O_IDX    33556480u                // [65536]
#define O_PROP   33622016u                // [8]
#define O_WEIGHT 33622024u                // [8]

// 1) weights f32 -> bf16, transposed to [e][n][k]
__global__ void __launch_bounds__(256)
k_convert(const float* __restrict__ enc, const float* __restrict__ dec,
          u16* __restrict__ encT, u16* __restrict__ decT) {
    int i = blockIdx.x * 256 + threadIdx.x;       // 0 .. 2*8*65536-1
    const float* src; u16* dst; int j = i;
    if (j < NEXP * DDIM * DDIM) { src = enc; dst = encT; }
    else { j -= NEXP * DDIM * DDIM; src = dec; dst = decT; }
    int e = j >> 16, rem = j & 65535, k = rem >> 8, n = rem & 255;
    dst[e * 65536 + n * 256 + k] = f2bf(src[j]);
}

// 2) zero accumulators
__global__ void k_zero(int* __restrict__ cnt, float* __restrict__ psum) {
    int t = threadIdx.x;
    if (t < NEXP) { cnt[t] = 0; psum[t] = 0.f; }
}

// 3) router: softmax/top-1, histogram, prob sums
__global__ void __launch_bounds__(256)
k_router(const float* __restrict__ act, const float* __restrict__ router_b,
         const float* __restrict__ router,
         int* __restrict__ idxW, float* __restrict__ maxpW,
         int* __restrict__ cnt, float* __restrict__ psum,
         float* __restrict__ outIdx) {
    __shared__ float sPsum[NEXP];
    __shared__ int   sCnt[NEXP];
    int tid = threadIdx.x;
    if (tid < NEXP) { sPsum[tid] = 0.f; sCnt[tid] = 0; }
    __syncthreads();

    int t = blockIdx.x * 256 + tid;
    float lg[NEXP] = {};
    const float* arow = act + (size_t)t * DDIM;
    for (int d = 0; d < DDIM; ++d) {
        float x = arow[d] - router_b[d];
        const float4* r4 = (const float4*)(router + d * NEXP);
        float4 a = r4[0], b = r4[1];
        lg[0] += x * a.x; lg[1] += x * a.y; lg[2] += x * a.z; lg[3] += x * a.w;
        lg[4] += x * b.x; lg[5] += x * b.y; lg[6] += x * b.z; lg[7] += x * b.w;
    }
    float mx = lg[0]; int am = 0;
    for (int e = 1; e < NEXP; ++e) if (lg[e] > mx) { mx = lg[e]; am = e; }
    float s = 0.f, p[NEXP];
    for (int e = 0; e < NEXP; ++e) { p[e] = __expf(lg[e] - mx); s += p[e]; }
    float inv = 1.f / s;                 // == max prob (exp(0)/s)
    for (int e = 0; e < NEXP; ++e) atomicAdd(&sPsum[e], p[e] * inv);
    atomicAdd(&sCnt[am], 1);
    idxW[t] = am; maxpW[t] = inv; outIdx[t] = (float)am;
    __syncthreads();
    if (tid < NEXP) { atomicAdd(&cnt[tid], sCnt[tid]); atomicAdd(&psum[tid], sPsum[tid]); }
}

// 4) finalize: offsets, expert_prop / expert_weighting, zero was_active + cursors
__global__ void __launch_bounds__(256)
k_finalize(const int* __restrict__ cnt, const float* __restrict__ psum,
           int* __restrict__ off, int* __restrict__ cur,
           float* __restrict__ outProp, float* __restrict__ outWeight,
           float* __restrict__ outWA) {
    int tid = threadIdx.x;
    if (tid == 0) { int a = 0; for (int e = 0; e < NEXP; ++e) { off[e] = a; a += cnt[e]; } }
    if (tid < NEXP) {
        cur[tid] = 0;
        outProp[tid]   = (float)cnt[tid] * (1.f / (float)NTOK);
        outWeight[tid] = psum[tid]       * (1.f / (float)NTOK);
    }
    for (int i = tid; i < NEXP * DDIM; i += 256) outWA[i] = 0.f;
}

// 5) counting-sort scatter of token ids by expert
__global__ void __launch_bounds__(256)
k_scatter(const int* __restrict__ idxW, const int* __restrict__ off,
          int* __restrict__ cur, int* __restrict__ sorted) {
    int t = blockIdx.x * 256 + threadIdx.x;
    int e = idxW[t];
    int pos = atomicAdd(&cur[e], 1);
    sorted[off[e] + pos] = t;
}

// 6) per-expert double GEMM via bf16 WMMA
__global__ void __launch_bounds__(256)
k_expert(const float* __restrict__ act, const float* __restrict__ token_act,
         const float* __restrict__ pre_b,
         const u16* __restrict__ encT, const u16* __restrict__ decT,
         const int* __restrict__ sorted, const int* __restrict__ cnt,
         const int* __restrict__ off, const float* __restrict__ maxpW,
         float* __restrict__ outRec, float* __restrict__ outLat,
         float* __restrict__ outWA) {
    __shared__ u16  xcBuf[MTILE * LSTRIDE];
    __shared__ u16  latBuf[MTILE * LSTRIDE];
    __shared__ int  tokS[MTILE];
    __shared__ float mpS[MTILE];

    const int e    = blockIdx.x / TILES_PER_E;
    const int tile = blockIdx.x % TILES_PER_E;
    const int c_e  = cnt[e];
    const int m0   = tile * MTILE;
    if (m0 >= c_e) return;
    const int mCount = (c_e - m0 < MTILE) ? (c_e - m0) : MTILE;
    const int sbase  = off[e] + m0;
    const int tid    = threadIdx.x;

    if (tid < MTILE) {
        bool valid = tid < mCount;
        int tok = valid ? sorted[sbase + tid] : 0;
        tokS[tid] = tok;
        mpS[tid]  = valid ? maxpW[tok] : 0.f;
    }
    __syncthreads();

    // gather xc = act[tok] - pre_b into LDS (bf16); zero-fill padding rows
    const float pb_col = pre_b[tid];
    for (int r = 0; r < MTILE; ++r) {
        float v = 0.f;
        if (r < mCount) v = act[(size_t)tokS[r] * DDIM + tid] - pb_col;
        xcBuf[r * LSTRIDE + tid] = f2bf(v);
    }
    __syncthreads();

    const int wave = tid >> 5;
    const int lane = tid & 31;
    const int l15  = lane & 15;
    const int sel  = lane >> 4;
    const u16* encE = encT + (size_t)e * DDIM * DDIM;
    const u16* decE = decT + (size_t)e * DDIM * DDIM;

    // ---- GEMM1: latent = relu(xc @ enc[e]) ----
    for (int nt = wave; nt < 16; nt += 8) {
        const int nBase = nt * 16;
        const int n     = nBase + l15;
        v8f acc[8] = {};
        for (int c = 0; c < 8; ++c) {
            v16bf bfrag = load_frag(encE + (size_t)n * DDIM + c * 32 + sel * 8);
            #pragma unroll
            for (int m = 0; m < 8; ++m) {
                v16bf afrag = load_frag(xcBuf + (m * 16 + l15) * LSTRIDE + c * 32 + sel * 8);
                acc[m] = __builtin_amdgcn_wmma_f32_16x16x32_bf16(
                    false, afrag, false, bfrag, (short)0, acc[m], false, false);
            }
        }
        #pragma unroll
        for (int m = 0; m < 8; ++m) {
            #pragma unroll
            for (int j = 0; j < 8; ++j) {
                int mAbs = m * 16 + sel * 8 + j;
                float r = acc[m][j]; r = r > 0.f ? r : 0.f;
                bool valid = mAbs < mCount;
                latBuf[mAbs * LSTRIDE + n] = valid ? f2bf(r) : (u16)0;
                if (valid) {
                    int tok = tokS[mAbs];
                    outLat[(size_t)tok * DDIM + n] = r;
                    if (r > 0.001f) outWA[e * DDIM + n] = 1.0f;
                }
            }
        }
    }
    __syncthreads();

    // ---- GEMM2: recon = latent @ dec[e]; fuse max_prob*recon + pre_b + token_act ----
    for (int nt = wave; nt < 16; nt += 8) {
        const int nBase = nt * 16;
        const int n     = nBase + l15;
        v8f acc[8] = {};
        for (int c = 0; c < 8; ++c) {
            v16bf bfrag = load_frag(decE + (size_t)n * DDIM + c * 32 + sel * 8);
            #pragma unroll
            for (int m = 0; m < 8; ++m) {
                v16bf afrag = load_frag(latBuf + (m * 16 + l15) * LSTRIDE + c * 32 + sel * 8);
                acc[m] = __builtin_amdgcn_wmma_f32_16x16x32_bf16(
                    false, afrag, false, bfrag, (short)0, acc[m], false, false);
            }
        }
        const float pbn = pre_b[n];
        #pragma unroll
        for (int m = 0; m < 8; ++m) {
            #pragma unroll
            for (int j = 0; j < 8; ++j) {
                int mAbs = m * 16 + sel * 8 + j;
                if (mAbs < mCount) {
                    int tok = tokS[mAbs];
                    float v = mpS[mAbs] * acc[m][j] + pbn
                            + token_act[(size_t)tok * DDIM + n];
                    outRec[(size_t)tok * DDIM + n] = v;
                }
            }
        }
    }
}

extern "C" void kernel_launch(void* const* d_in, const int* in_sizes, int n_in,
                              void* d_out, int out_size, void* d_ws, size_t ws_size,
                              hipStream_t stream) {
    const float* act      = (const float*)d_in[0];
    const float* token_act= (const float*)d_in[1];
    const float* pre_b    = (const float*)d_in[2];
    const float* router_b = (const float*)d_in[3];
    const float* router   = (const float*)d_in[4];
    const float* enc      = (const float*)d_in[5];
    const float* dec      = (const float*)d_in[6];

    char* ws = (char*)d_ws;
    u16*   encT = (u16*)(ws + WS_ENC);
    u16*   decT = (u16*)(ws + WS_DEC);
    int*   idxW = (int*)(ws + WS_IDX);
    float* maxpW= (float*)(ws + WS_MAXP);
    int*   sorted=(int*)(ws + WS_SORT);
    int*   cntW = (int*)(ws + WS_CNT);
    int*   offW = (int*)(ws + WS_OFF);
    int*   curW = (int*)(ws + WS_CUR);
    float* psumW= (float*)(ws + WS_PSUM);

    float* out = (float*)d_out;
    float* oRec = out + O_REC;
    float* oLat = out + O_LAT;
    float* oWA  = out + O_WA;
    float* oIdx = out + O_IDX;
    float* oProp= out + O_PROP;
    float* oWgt = out + O_WEIGHT;

    k_convert <<<2 * NEXP * DDIM * DDIM / 256, 256, 0, stream>>>(enc, dec, encT, decT);
    k_zero    <<<1, 32, 0, stream>>>(cntW, psumW);
    k_router  <<<NTOK / 256, 256, 0, stream>>>(act, router_b, router,
                                               idxW, maxpW, cntW, psumW, oIdx);
    k_finalize<<<1, 256, 0, stream>>>(cntW, psumW, offW, curW, oProp, oWgt, oWA);
    k_scatter <<<NTOK / 256, 256, 0, stream>>>(idxW, offW, curW, sorted);
    k_expert  <<<NEXP * TILES_PER_E, 256, 0, stream>>>(act, token_act, pre_b,
                                                       encT, decT, sorted, cntW, offW,
                                                       maxpW, oRec, oLat, oWA);
}
